// SA_Layer_68719477127
// MI455X (gfx1250) — matrix operand order repaired and checked by reading, not compiled, not run on hardware
//
#include <hip/hip_runtime.h>
#include <hip/hip_bf16.h>
#include <math.h>

// Problem constants (B=4, C=256, N=4096, C4=64)
#define BB 4
#define CC 256
#define NN 4096
#define C4 64

typedef __attribute__((ext_vector_type(16))) __bf16 bf16x16;
typedef __attribute__((ext_vector_type(8)))  float   floatx8;

union FragU {
    bf16x16 v;
    uint4   q[2];
};

union Bf8U {
    uint4  q;
    __bf16 h[8];
};

__device__ __forceinline__ floatx8 wmma_bf16f32(const FragU& a, const FragU& b, floatx8 c) {
    return __builtin_amdgcn_wmma_f32_16x16x32_bf16(false, a.v, false, b.v, (short)0, c, false, false);
}

// A-matrix fragment (16x32 bf16): A stored row-major, lda in elements.
// lane l: m = l&15, hi = l>>4; element j -> k = (j&15) + 16*hi
__device__ __forceinline__ FragU load_a_frag_g(const __bf16* A, int lda, int row, int k0, int hi) {
    FragU f;
    const uint4* p = reinterpret_cast<const uint4*>(A + (size_t)row * lda + k0 + hi * 16);
    f.q[0] = p[0];   // k0 + 16*hi + 0..7
    f.q[1] = p[1];   // k0 + 16*hi + 8..15
    return f;
}

// B-matrix fragment (32x16 bf16): element (k,col) stored at B[col*ldb + k] (k-contiguous).
// lane l: col = l&15, hi = l>>4; element j -> k = (j&15) + 16*hi
__device__ __forceinline__ FragU load_b_frag_g(const __bf16* Bm, int ldb, int col, int k0, int hi) {
    FragU f;
    const uint4* p = reinterpret_cast<const uint4*>(Bm + (size_t)col * ldb + k0 + hi * 16);
    f.q[0] = p[0];
    f.q[1] = p[1];
    return f;
}

// Generic K-loop 16x16 tile: D = A(rowbase:+16, 0:K) * B(0:K, colbase:+16)
__device__ __forceinline__ floatx8 gemm_tile(const __bf16* A, int lda, int rowbase,
                                             const __bf16* Bm, int ldb, int colbase, int K) {
    int lane = threadIdx.x & 31;
    int m16  = lane & 15;
    int hi   = lane >> 4;
    floatx8 acc;
#pragma unroll
    for (int i = 0; i < 8; ++i) acc[i] = 0.f;
    for (int k0 = 0; k0 < K; k0 += 32) {
        FragU a = load_a_frag_g(A, lda, rowbase + m16, k0, hi);
        FragU b = load_b_frag_g(Bm, ldb, colbase + m16, k0, hi);
        acc = wmma_bf16f32(a, b, acc);
    }
    return acc;
}

// ---------------------------------------------------------------------------
// K0a: x [B][C][N] f32 -> xT [B][N][C] bf16
__global__ __launch_bounds__(256) void k_convert_xT(const float* __restrict__ x,
                                                    __bf16* __restrict__ xT) {
    size_t idx = (size_t)blockIdx.x * blockDim.x + threadIdx.x;
    if (idx >= (size_t)BB * CC * NN) return;
    size_t b   = idx / ((size_t)CC * NN);
    size_t rem = idx % ((size_t)CC * NN);
    size_t c   = rem / NN;
    size_t n   = rem % NN;
    xT[(b * NN + n) * CC + c] = (__bf16)x[idx];
}

// K0b: generic f32 -> bf16
__global__ __launch_bounds__(256) void k_convert(const float* __restrict__ s,
                                                 __bf16* __restrict__ d, int n) {
    int i = blockIdx.x * blockDim.x + threadIdx.x;
    if (i < n) d[i] = (__bf16)s[i];
}

// ---------------------------------------------------------------------------
// K1: Xq[b][n][q] = sum_c xT[b][n][c] * w_qk[q][c]   (A = xT, B = w_qk)
__global__ __launch_bounds__(128) void k_xq(const __bf16* __restrict__ xT,
                                            const __bf16* __restrict__ wqk,
                                            __bf16* __restrict__ Xq) {
    int w    = blockIdx.x * 4 + (threadIdx.x >> 5);   // 4096 tiles total
    int b    = w >> 10;
    int t    = w & 1023;
    int nt   = t >> 2;        // 0..255
    int qt   = t & 3;         // 0..3
    int lane = threadIdx.x & 31;
    int m16  = lane & 15;
    int hi   = lane >> 4;

    const __bf16* A = xT + (size_t)b * NN * CC;
    floatx8 acc = gemm_tile(A, CC, nt * 16, wqk, CC, qt * 16, CC);

    int q = qt * 16 + m16;
#pragma unroll
    for (int r = 0; r < 8; ++r) {
        int n = nt * 16 + hi * 8 + r;
        Xq[((size_t)b * NN + n) * C4 + q] = (__bf16)acc[r];
    }
}

// K2: Xv[b][c][n] = sum_k w_v[c][k] * xT[b][n][k] + b_v[c]
__global__ __launch_bounds__(128) void k_xv(const __bf16* __restrict__ wv,
                                            const __bf16* __restrict__ xT,
                                            const float* __restrict__ bv,
                                            __bf16* __restrict__ Xv) {
    int w    = blockIdx.x * 4 + (threadIdx.x >> 5);   // 16384 tiles total
    int b    = w >> 12;
    int t    = w & 4095;
    int ct   = t >> 8;        // 0..15
    int nt   = t & 255;       // 0..255
    int lane = threadIdx.x & 31;
    int m16  = lane & 15;
    int hi   = lane >> 4;

    const __bf16* Bm = xT + (size_t)b * NN * CC;
    floatx8 acc = gemm_tile(wv, CC, ct * 16, Bm, CC, nt * 16, CC);

    int n = nt * 16 + m16;
#pragma unroll
    for (int r = 0; r < 8; ++r) {
        int c = ct * 16 + hi * 8 + r;
        Xv[((size_t)b * CC + c) * NN + n] = (__bf16)(acc[r] + bv[c]);
    }
}

// ---------------------------------------------------------------------------
// K3: per-row online softmax stats over energy = Xq * Xq^T
// rmax[b][n] = rowmax, rsinv[b][n] = 1 / sum_m exp(e[n][m] - rowmax[n])
__global__ __launch_bounds__(128) void k_stats(const __bf16* __restrict__ Xq,
                                               float* __restrict__ rmax,
                                               float* __restrict__ rsinv) {
    int w    = blockIdx.x * 4 + (threadIdx.x >> 5);   // 1024 waves total
    int b    = w >> 8;
    int nt   = w & 255;
    int lane = threadIdx.x & 31;
    int m16  = lane & 15;
    int hi   = lane >> 4;

    const __bf16* Xq_b = Xq + (size_t)b * NN * C4;
    // A fragments (rows = n strip), K=64 -> 2 steps, kept in registers
    FragU a0 = load_a_frag_g(Xq_b, C4, nt * 16 + m16, 0, hi);
    FragU a1 = load_a_frag_g(Xq_b, C4, nt * 16 + m16, 32, hi);

    float runmax[8], runsum[8];
#pragma unroll
    for (int r = 0; r < 8; ++r) { runmax[r] = -INFINITY; runsum[r] = 0.f; }

    for (int mt = 0; mt < NN / 16; ++mt) {
        FragU b0 = load_b_frag_g(Xq_b, C4, mt * 16 + m16, 0, hi);
        FragU b1 = load_b_frag_g(Xq_b, C4, mt * 16 + m16, 32, hi);
        floatx8 e;
#pragma unroll
        for (int i = 0; i < 8; ++i) e[i] = 0.f;
        e = wmma_bf16f32(a0, b0, e);
        e = wmma_bf16f32(a1, b1, e);
#pragma unroll
        for (int r = 0; r < 8; ++r) {
            float v = e[r];
            if (v > runmax[r]) {
                runsum[r] = runsum[r] * __expf(runmax[r] - v) + 1.f;
                runmax[r] = v;
            } else {
                runsum[r] += __expf(v - runmax[r]);
            }
        }
    }
    // merge across the 16 lanes holding different columns of the same rows
#pragma unroll
    for (int off = 1; off < 16; off <<= 1) {
#pragma unroll
        for (int r = 0; r < 8; ++r) {
            float om = __shfl_xor(runmax[r], off, 16);
            float os = __shfl_xor(runsum[r], off, 16);
            float nm = fmaxf(runmax[r], om);
            runsum[r] = runsum[r] * __expf(runmax[r] - nm) + os * __expf(om - nm);
            runmax[r] = nm;
        }
    }
    if (m16 == 0) {
#pragma unroll
        for (int r = 0; r < 8; ++r) {
            int n = nt * 16 + hi * 8 + r;
            rmax[(size_t)b * NN + n]  = runmax[r];
            rsinv[(size_t)b * NN + n] = 1.0f / runsum[r];   // runsum >= 1 always
        }
    }
}

// ---------------------------------------------------------------------------
// K4: fused energy-recompute + exp + x_r' = Xv*P + colsum + d = x - x_r.
// One block owns 32 columns (m) of one batch; loops over all n in steps of 64.
// Stage 1: all 8 waves each compute one 16x16 energy tile (2x4 grid), apply
//          exp(e - rowmax) * rsinv, store P [32m][64n] bf16 to LDS.
// Stage 2: each wave: 4 accumulator tiles (64 v-rows x 16 m-cols), K=64.
// Writes dT[b][n][c] bf16 (ready as B-operand of the w_t GEMM).
__global__ __launch_bounds__(256) void k_xr_d(const __bf16* __restrict__ Xq,
                                              const __bf16* __restrict__ Xv,
                                              const __bf16* __restrict__ xT,
                                              const float* __restrict__ rmax,
                                              const float* __restrict__ rsinv,
                                              __bf16* __restrict__ dT) {
    __shared__ __align__(16) __bf16 Plds[32 * 64];   // [m_local][n_local], 4 KB
    __shared__ float csum_lds[32];

    const int b    = blockIdx.y;
    const int mb   = blockIdx.x;                     // 0..127 -> 32 columns each
    const int tid  = threadIdx.x;
    const int wave = tid >> 5;
    const int lane = tid & 31;
    const int m16  = lane & 15;
    const int hi   = lane >> 4;

    const __bf16* Xq_b = Xq + (size_t)b * NN * C4;
    const __bf16* Xv_b = Xv + (size_t)b * CC * NN;
    const float*  rm_b = rmax + (size_t)b * NN;
    const float*  rs_b = rsinv + (size_t)b * NN;

    if (tid < 32) csum_lds[tid] = 0.f;
    __syncthreads();

    const int nsub     = wave & 3;    // stage-1: n sub-tile (0..3)
    const int emtile   = wave >> 2;   // stage-1: m sub-tile (0..1)
    const int s2_mtile = wave & 1;    // stage-2: m sub-tile
    const int vt_base  = (wave >> 1) * 4;

    floatx8 acc[4];
#pragma unroll
    for (int i = 0; i < 4; ++i)
#pragma unroll
        for (int r = 0; r < 8; ++r) acc[i][r] = 0.f;
    float csum = 0.f;

    for (int n0 = 0; n0 < NN; n0 += 64) {
        // ---- stage 1: energy tile -> P (softmax numerator * rsinv) in LDS
        {
            int rowbase = n0 + nsub * 16;
            int colbase = mb * 32 + emtile * 16;
            FragU ea0 = load_a_frag_g(Xq_b, C4, rowbase + m16, 0, hi);
            FragU ea1 = load_a_frag_g(Xq_b, C4, rowbase + m16, 32, hi);
            FragU eb0 = load_b_frag_g(Xq_b, C4, colbase + m16, 0, hi);
            FragU eb1 = load_b_frag_g(Xq_b, C4, colbase + m16, 32, hi);
            floatx8 e;
#pragma unroll
            for (int i = 0; i < 8; ++i) e[i] = 0.f;
            e = wmma_bf16f32(ea0, eb0, e);
            e = wmma_bf16f32(ea1, eb1, e);

            __bf16* prow = &Plds[(emtile * 16 + m16) * 64 + nsub * 16 + hi * 8];
#pragma unroll
            for (int r = 0; r < 8; ++r) {
                int n   = rowbase + hi * 8 + r;
                float p = __expf(e[r] - rm_b[n]) * rs_b[n];
                csum += p;
                prow[r] = (__bf16)p;
            }
        }
        __syncthreads();
        // ---- stage 2: acc += Xv(:, n0:n0+64) * P   (K=64 -> 2 chained WMMAs)
#pragma unroll
        for (int ks = 0; ks < 2; ++ks) {
            int k0 = ks * 32;
            FragU bf;
            const uint4* bp =
                reinterpret_cast<const uint4*>(&Plds[(s2_mtile * 16 + m16) * 64 + k0 + hi * 16]);
            bf.q[0] = bp[0];
            bf.q[1] = bp[1];
#pragma unroll
            for (int i = 0; i < 4; ++i) {
                int row = (vt_base + i) * 16 + m16;
                FragU af = load_a_frag_g(Xv_b, NN, row, n0 + k0, hi);
                if (ks == 0 && i == 0 && n0 + 64 < NN) {
                    // pull next Xv panel toward the WGP (global_prefetch_b8)
                    __builtin_prefetch(Xv_b + (size_t)row * NN + n0 + 64, 0, 3);
                }
                acc[i] = wmma_bf16f32(af, bf, acc[i]);
            }
        }
        __syncthreads();
    }

    // ---- column-sum reduction (every wave holds partials for its emtile)
    {
        float c2 = csum + __shfl_xor(csum, 16);
        if (lane < 16) atomicAdd(&csum_lds[emtile * 16 + lane], c2);
    }
    __syncthreads();

    // ---- finalize: x_r = acc / colsum ; d = x - x_r -> dT[b][n][c] bf16
    const int   nglob = mb * 32 + s2_mtile * 16 + m16;
    const float rinv  = 1.0f / (1e-9f + csum_lds[s2_mtile * 16 + m16]);
    const size_t rowo = (size_t)b * NN * CC + (size_t)nglob * CC;
#pragma unroll
    for (int i = 0; i < 4; ++i) {
        int cbase = (vt_base + i) * 16 + hi * 8;
        Bf8U xx, dd;
        xx.q = *reinterpret_cast<const uint4*>(xT + rowo + cbase);
#pragma unroll
        for (int r = 0; r < 8; ++r) {
            float xv = (float)xx.h[r];
            dd.h[r]  = (__bf16)(xv - acc[i][r] * rinv);
        }
        *reinterpret_cast<uint4*>(dT + rowo + cbase) = dd.q;
    }
}

// ---------------------------------------------------------------------------
// K5: y = BN(w_t * d + b_t); out = x + relu(y)
__global__ __launch_bounds__(128) void k_out(const __bf16* __restrict__ wt,
                                             const __bf16* __restrict__ dT,
                                             const float* __restrict__ bt,
                                             const float* __restrict__ gamma,
                                             const float* __restrict__ beta,
                                             const float* __restrict__ mean,
                                             const float* __restrict__ var,
                                             const float* __restrict__ x,
                                             float* __restrict__ out) {
    int w    = blockIdx.x * 4 + (threadIdx.x >> 5);   // 16384 tiles total
    int b    = w >> 12;
    int t    = w & 4095;
    int ct   = t >> 8;
    int nt   = t & 255;
    int lane = threadIdx.x & 31;
    int m16  = lane & 15;
    int hi   = lane >> 4;

    const __bf16* Bm = dT + (size_t)b * NN * CC;
    floatx8 acc = gemm_tile(wt, CC, ct * 16, Bm, CC, nt * 16, CC);

    int n = nt * 16 + m16;
#pragma unroll
    for (int r = 0; r < 8; ++r) {
        int c     = ct * 16 + hi * 8 + r;
        float y   = acc[r] + bt[c];
        y         = gamma[c] * (y - mean[c]) * rsqrtf(var[c] + 1e-5f) + beta[c];
        size_t xi = ((size_t)b * CC + c) * NN + n;
        out[xi]   = x[xi] + fmaxf(y, 0.f);
    }
}

// ---------------------------------------------------------------------------
extern "C" void kernel_launch(void* const* d_in, const int* in_sizes, int n_in,
                              void* d_out, int out_size, void* d_ws, size_t ws_size,
                              hipStream_t stream) {
    const float* x     = (const float*)d_in[0];
    const float* w_qk  = (const float*)d_in[1];
    const float* w_v   = (const float*)d_in[2];
    const float* b_v   = (const float*)d_in[3];
    const float* w_t   = (const float*)d_in[4];
    const float* b_t   = (const float*)d_in[5];
    const float* gamma = (const float*)d_in[6];
    const float* beta  = (const float*)d_in[7];
    const float* mean  = (const float*)d_in[8];
    const float* var   = (const float*)d_in[9];
    float* out = (float*)d_out;

    char* ws = (char*)d_ws;
    size_t off = 0;
    __bf16* xT    = (__bf16*)(ws + off); off += (size_t)BB * NN * CC * 2;   // 8 MB
    __bf16* wqkb  = (__bf16*)(ws + off); off += (size_t)C4 * CC * 2;        // 32 KB
    __bf16* wvb   = (__bf16*)(ws + off); off += (size_t)CC * CC * 2;        // 128 KB
    __bf16* wtb   = (__bf16*)(ws + off); off += (size_t)CC * CC * 2;        // 128 KB
    __bf16* Xq    = (__bf16*)(ws + off); off += (size_t)BB * NN * C4 * 2;   // 2 MB
    __bf16* Xv    = (__bf16*)(ws + off); off += (size_t)BB * CC * NN * 2;   // 8 MB
    float*  rmax  = (float*)(ws + off);  off += (size_t)BB * NN * 4;        // 64 KB
    float*  rsinv = (float*)(ws + off);  off += (size_t)BB * NN * 4;        // 64 KB
    __bf16* dT    = (__bf16*)(ws + off); off += (size_t)BB * NN * CC * 2;   // 8 MB

    // 0) precision conversion / transpose
    k_convert_xT<<<(BB * CC * NN) / 256, 256, 0, stream>>>(x, xT);
    k_convert<<<(C4 * CC + 255) / 256, 256, 0, stream>>>(w_qk, wqkb, C4 * CC);
    k_convert<<<(CC * CC + 255) / 256, 256, 0, stream>>>(w_v, wvb, CC * CC);
    k_convert<<<(CC * CC + 255) / 256, 256, 0, stream>>>(w_t, wtb, CC * CC);

    // 1) Xq = xT * w_qk^T        (4096 tile-waves)
    k_xq<<<1024, 128, 0, stream>>>(xT, wqkb, Xq);
    // 2) Xv = w_v * x + b_v      (16384 tile-waves)
    k_xv<<<4096, 128, 0, stream>>>(wvb, xT, b_v, Xv);
    // 3) row softmax stats       (1024 waves)
    k_stats<<<256, 128, 0, stream>>>(Xq, rmax, rsinv);
    // 4) fused attention apply   (128 m-blocks x 4 batches)
    k_xr_d<<<dim3(128, BB), 256, 0, stream>>>(Xq, Xv, xT, rmax, rsinv, dT);
    // 5) output GEMM + BN + relu + residual (16384 tile-waves)
    k_out<<<4096, 128, 0, stream>>>(wtb, dT, b_t, gamma, beta, mean, var, x, out);
}
